// FormulaAI_36429912605019
// MI455X (gfx1250) — matrix operand order repaired and checked by reading, not compiled, not run on hardware
//
#include <hip/hip_runtime.h>
#include <cmath>

// ---------------------------------------------------------------------------
// Model dimensions (from the reference)
// ---------------------------------------------------------------------------
#define Bc    2
#define TIMG  800
#define DPIC  400
#define DIMG  512
#define TTXT  189
#define DTXT  128
#define NHV   64
#define HSV   256
#define NHL   32
#define HSL   128
#define NHC   32
#define HSC   256
#define DCR   512
#define VOC   304

// ---------------------------------------------------------------------------
// WMMA plumbing (gfx1250, wave32): f16 inputs, f32 accumulate
// ---------------------------------------------------------------------------
typedef __attribute__((ext_vector_type(16))) _Float16 v16h;
typedef __attribute__((ext_vector_type(8)))  _Float16 v8h;
typedef __attribute__((ext_vector_type(8)))  float    v8f;
typedef __attribute__((ext_vector_type(4)))  float    v4f;

union AFrag { v16h v; _Float16 h[16]; };

__device__ __forceinline__ v8f wmma_f16(v16h a, v16h b, v8f c) {
    // D = A(16x32) * B(32x16) + C(16x16), f32 accumulate
    return __builtin_amdgcn_wmma_f32_16x16x32_f16(false, a, false, b,
                                                  (short)0, c, false, false);
}

// two contiguous 8-half (16B) LDS loads -> one 16-half fragment
__device__ __forceinline__ v16h ld2x8(const _Float16* p0, const _Float16* p1) {
    v8h a = *(const v8h*)p0;
    v8h b = *(const v8h*)p1;
    return __builtin_shufflevector(a, b, 0, 1, 2, 3, 4, 5, 6, 7,
                                         8, 9, 10, 11, 12, 13, 14, 15);
}
// one contiguous 16-half (32B) LDS load pair
__device__ __forceinline__ v16h ld16(const _Float16* p) { return ld2x8(p, p + 8); }

// ---------------------------------------------------------------------------
// CDNA5 async copy: 16B per lane, global -> LDS, tracked by ASYNCcnt.
// LDS address = low 32 bits of the generic shared pointer (LDS aperture).
// ---------------------------------------------------------------------------
__device__ __forceinline__ void async_ld_b128(unsigned ldsoff, const _Float16* g) {
    asm volatile("global_load_async_to_lds_b128 %0, %1, off"
                 :: "v"(ldsoff), "v"(g) : "memory");
}
__device__ __forceinline__ void wait_async0() {
    asm volatile("s_wait_asynccnt 0x0" ::: "memory");
}

// ---------------------------------------------------------------------------
// Generic GEMM: C[M,N] = relu?( A[M,K] @ Bt^T + bias + Res )
//   A : f32 [M,K] activations (converted to f16 while staging in LDS)
//   Bt: f16 [Np,Kp] PRE-TRANSPOSED padded weights; B tile DMA'd into LDS
//       with global_load_async_to_lds_b128 (no conversion, no bounds checks).
// Batched via grid.z.
// ---------------------------------------------------------------------------
#define GT_M 64
#define GT_N 64
#define GT_K 32

__global__ void __launch_bounds__(256)
gemm_wmma_k(const float* __restrict__ A, int lda, long long sA,
            const _Float16* __restrict__ Bt, int kp, long long sB,
            const float* __restrict__ bias,
            const float* __restrict__ Res, int ldres,
            float* __restrict__ C, int ldc, long long sC,
            int M, int N, int K, int relu)
{
    __shared__ _Float16 As[GT_M][GT_K];     // (m, k)
    __shared__ _Float16 BsT[GT_N][GT_K];    // (n, k)  transposed weight tile

    const int bz = blockIdx.z;
    A  += (long long)bz * sA;
    Bt += (long long)bz * sB;
    C  += (long long)bz * sC;

    const int tid  = threadIdx.x;
    const int wid  = tid >> 5;
    const int lane = tid & 31;
    const int m0 = blockIdx.y * GT_M;
    const int n0 = blockIdx.x * GT_N;
    const int wm = (wid >> 1) * 16;     // wave row offset in macro tile
    const int wn = (wid & 1) * 32;      // wave col offset in macro tile

    // staging coordinates (8 elements per thread per tile)
    const int ar = tid >> 2;            // 0..63  (A row / B row n)
    const int ac = (tid & 3) * 8;       // 0..24  (col base, elements)

    const bool mfull = (m0 + GT_M <= M);

    // hoisted async-copy source/dest for the B tile (always in-bounds: padded)
    const unsigned  bdst = (unsigned)(unsigned long long)&BsT[ar][ac];
    const _Float16* bsrc = Bt + (long long)(n0 + ar) * kp + ac;

    v8f acc0 = {};
    v8f acc1 = {};

    for (int kc = 0; kc < K; kc += GT_K) {
        const bool kfull = (kc + GT_K <= K);

        // ---- B tile: async DMA straight into LDS (ASYNCcnt) ----
        async_ld_b128(bdst, bsrc + kc);

        // ---- A tile: f32 -> f16 while staging ----
        if (mfull && kfull) {
            const float* ap = A + (long long)(m0 + ar) * lda + kc + ac;
            v4f f0 = *(const v4f*)ap;
            v4f f1 = *(const v4f*)(ap + 4);
            if (kc + GT_K < K)
                __builtin_prefetch(ap + GT_K, 0, 1);
#pragma unroll
            for (int j = 0; j < 4; ++j) {
                As[ar][ac + j]     = (_Float16)f0[j];
                As[ar][ac + 4 + j] = (_Float16)f1[j];
            }
        } else {
            float v[8];
#pragma unroll
            for (int j = 0; j < 8; ++j) {
                int gm = m0 + ar, gk = kc + ac + j;
                int gmc = gm < M ? gm : M - 1;
                int gkc = gk < K ? gk : K - 1;
                v[j] = A[(long long)gmc * lda + gkc];
                if (gm >= M || gk >= K) v[j] = 0.f;
            }
#pragma unroll
            for (int j = 0; j < 8; ++j) As[ar][ac + j] = (_Float16)v[j];
        }

        wait_async0();          // our async copies have landed in LDS
        __syncthreads();        // everyone's copies + A stores visible

        // ---- fragments (all contiguous per-lane ds_load_b128) ----
        AFrag a, b0, b1;
        {
            const int m  = wm + (lane & 15);
            const int kb = (lane < 16) ? 0 : 16;
            a.v = ld16(&As[m][kb]);
        }
        {
            const int kb = (lane < 16) ? 0 : 16;
            const int nl = wn + (lane & 15);
            b0.v = ld16(&BsT[nl][kb]);
            b1.v = ld16(&BsT[nl + 16][kb]);
        }
        acc0 = wmma_f16(a.v, b0.v, acc0);
        acc1 = wmma_f16(a.v, b1.v, acc1);
        __syncthreads();
    }

    // ---- epilogue: C/D layout — col = lane&15, row = vgpr + 8*(lane>>4) ----
    const int colA = n0 + wn + (lane & 15);
    const int colB = colA + 16;
    const int rb   = 8 * (lane >> 4);
#pragma unroll
    for (int r = 0; r < 8; ++r) {
        int row = m0 + wm + rb + r;
        if (row >= M) continue;
        long long crow = (long long)row * ldc;
        if (colA < N) {
            float v = acc0[r];
            if (bias) v += bias[colA];
            if (Res)  v += Res[(long long)row * ldres + colA];
            if (relu) v = fmaxf(v, 0.f);
            C[crow + colA] = v;
        }
        if (colB < N) {
            float v = acc1[r];
            if (bias) v += bias[colB];
            if (Res)  v += Res[(long long)row * ldres + colB];
            if (relu) v = fmaxf(v, 0.f);
            C[crow + colB] = v;
        }
    }
}

// ---------------------------------------------------------------------------
// Weight convert + transpose: W[K,N] f32 -> Wt[Np,Kp] f16, zero padded.
// Batched over grid.z (per-head weight matrices are contiguous).
// ---------------------------------------------------------------------------
__global__ void wcvt_k(const float* __restrict__ W, _Float16* __restrict__ Wt,
                       int K, int N, int Kp, int Np)
{
    const long long bz = blockIdx.z;
    W  += bz * (long long)K * N;
    Wt += bz * (long long)Np * Kp;
    const long long tot = (long long)Np * Kp;
    for (long long i = blockIdx.x * (long long)blockDim.x + threadIdx.x;
         i < tot; i += (long long)gridDim.x * blockDim.x) {
        int n = (int)(i / Kp);
        int k = (int)(i - (long long)n * Kp);
        float v = (n < N && k < K) ? W[(long long)k * N + n] : 0.f;
        Wt[i] = (_Float16)v;
    }
}

// ---------------------------------------------------------------------------
// Attention: one workgroup = (b, head, 16-query tile).
// Q/K/V stored as [H][B][T][HS] f32. O written merged as [B][Tq][H*HS].
// ---------------------------------------------------------------------------
#define ATT_MAXTK 800   // multiple of 32

__global__ void __launch_bounds__(256)
attn_wmma_k(const float* __restrict__ Q, const float* __restrict__ Kg,
            const float* __restrict__ Vg, float* __restrict__ O,
            const int* __restrict__ maskp,
            int Bn, int H, int Tq, int Tk, int HS, float scale, int causal)
{
    __shared__ _Float16 Qs[16 * 256];          // Q tile, f16         (8 KB)
    __shared__ _Float16 SP[16 * ATT_MAXTK];    // scores/probs, f16   (25.6 KB)
    __shared__ _Float16 Vs[256 * 32];          // V chunk (n,k), f16  (16 KB)
    __shared__ float    red2[16 * 16];         // softmax reductions  (1 KB)

    const int q0   = blockIdx.x * 16;
    const int h    = blockIdx.y;
    const int b    = blockIdx.z;
    const int tid  = threadIdx.x;
    const int wid  = tid >> 5;
    const int lane = tid & 31;
    const int Tkpad = (Tk + 31) & ~31;

    const float* Qh = Q  + ((long long)(h * Bn + b) * Tq) * HS;
    const float* Kh = Kg + ((long long)(h * Bn + b) * Tk) * HS;
    const float* Vh = Vg + ((long long)(h * Bn + b) * Tk) * HS;

    // ---- stage Q tile (vectorized; rows past Tq -> 0) ----
    {
        const int qr  = tid >> 4;                // 0..15
        const int cw  = HS >> 4;                 // floats per thread (8 or 16)
        const int qc0 = (tid & 15) * cw;
        const int qrow = q0 + qr;
        const float* qp = Qh + (long long)qrow * HS + qc0;
        for (int c = 0; c < cw; c += 4) {
            v4f f = {};
            if (qrow < Tq) f = *(const v4f*)(qp + c);
#pragma unroll
            for (int j = 0; j < 4; ++j)
                Qs[qr * HS + qc0 + c + j] = (_Float16)f[j];
        }
    }
    __syncthreads();

    // ---- hoist Q A-fragments (per wave, contiguous LDS reads) ----
    const int nkc = HS / 32;
    AFrag qa[8];
    {
        const int m  = lane & 15;
        const int kb = (lane < 16) ? 0 : 16;
        for (int kc = 0; kc < nkc; ++kc)
            qa[kc].v = ld16(&Qs[m * HS + kc * 32 + kb]);
    }

    // ---- phase 1: scores (K fragments = contiguous global b128 loads) ----
    for (int sc = wid * 32; sc < Tkpad; sc += 8 * 32) {
        v8f acc0 = {};
        v8f acc1 = {};
        const int nn = lane & 15;
        const int s0 = sc + nn;
        const int s1 = sc + 16 + nn;
        const int kb = (lane < 16) ? 0 : 16;
        for (int kc = 0; kc < nkc; ++kc) {
            AFrag b0, b1;
            const int koff = kc * 32 + kb;
            if (s0 < Tk) {
                const float* p = Kh + (long long)s0 * HS + koff;
                v4f f0 = *(const v4f*)p;
                v4f f1 = *(const v4f*)(p + 4);
                v4f f2 = *(const v4f*)(p + 8);
                v4f f3 = *(const v4f*)(p + 12);
#pragma unroll
                for (int j = 0; j < 4; ++j) {
                    b0.h[j]      = (_Float16)f0[j];
                    b0.h[4 + j]  = (_Float16)f1[j];
                    b0.h[8 + j]  = (_Float16)f2[j];
                    b0.h[12 + j] = (_Float16)f3[j];
                }
            } else {
#pragma unroll
                for (int j = 0; j < 16; ++j) b0.h[j] = (_Float16)0.f;
            }
            if (s1 < Tk) {
                const float* p = Kh + (long long)s1 * HS + koff;
                v4f f0 = *(const v4f*)p;
                v4f f1 = *(const v4f*)(p + 4);
                v4f f2 = *(const v4f*)(p + 8);
                v4f f3 = *(const v4f*)(p + 12);
#pragma unroll
                for (int j = 0; j < 4; ++j) {
                    b1.h[j]      = (_Float16)f0[j];
                    b1.h[4 + j]  = (_Float16)f1[j];
                    b1.h[8 + j]  = (_Float16)f2[j];
                    b1.h[12 + j] = (_Float16)f3[j];
                }
            } else {
#pragma unroll
                for (int j = 0; j < 16; ++j) b1.h[j] = (_Float16)0.f;
            }
            acc0 = wmma_f16(qa[kc].v, b0.v, acc0);
            acc1 = wmma_f16(qa[kc].v, b1.v, acc1);
        }
        const int rb = 8 * (lane >> 4);
#pragma unroll
        for (int r = 0; r < 8; ++r) {
            SP[(rb + r) * Tkpad + sc + nn]      = (_Float16)(acc0[r] * scale);
            SP[(rb + r) * Tkpad + sc + 16 + nn] = (_Float16)(acc1[r] * scale);
        }
    }
    __syncthreads();

    // ---- phase 2: masked softmax (16 threads per query row) ----
    {
        const int r   = tid >> 4;
        const int sub = tid & 15;
        const int qrow = q0 + r;
        int slim = 0;
        if (qrow < Tq) {
            slim = Tk;
            if (causal) { slim = qrow + 1; if (slim > Tk) slim = Tk; }
        }
        float mx = -1e30f;
        for (int s = sub; s < slim; s += 16)
            if (!maskp || maskp[b * Tk + s] != 0)
                mx = fmaxf(mx, (float)SP[r * Tkpad + s]);
        red2[r * 16 + sub] = mx;
        __syncthreads();
        for (int o = 8; o > 0; o >>= 1) {
            if (sub < o)
                red2[r * 16 + sub] = fmaxf(red2[r * 16 + sub], red2[r * 16 + sub + o]);
            __syncthreads();
        }
        mx = red2[r * 16];
        __syncthreads();

        float sum = 0.f;
        for (int s = sub; s < Tkpad; s += 16) {
            float p = 0.f;
            if (s < slim && (!maskp || maskp[b * Tk + s] != 0)) {
                p = __expf((float)SP[r * Tkpad + s] - mx);
                sum += p;
            }
            SP[r * Tkpad + s] = (_Float16)p;
        }
        red2[r * 16 + sub] = sum;
        __syncthreads();
        for (int o = 8; o > 0; o >>= 1) {
            if (sub < o)
                red2[r * 16 + sub] += red2[r * 16 + sub + o];
            __syncthreads();
        }
        sum = red2[r * 16];
        __syncthreads();

        const float rinv = (sum > 0.f) ? (1.f / sum) : 0.f;
        for (int s = sub; s < Tkpad; s += 16)
            SP[r * Tkpad + s] = (_Float16)((float)SP[r * Tkpad + s] * rinv);
    }
    __syncthreads();

    // ---- phase 3: O = P @ V (V staged transposed in LDS per 32-key chunk) ----
    const int ntiles = HS / 16;           // 8 or 16; waves own nt = wid (+8)
    v8f oacc0 = {};
    v8f oacc1 = {};
    const int kbA = (lane < 16) ? 0 : 16;
    const int kbB = (lane < 16) ? 0 : 16;
    for (int sc = 0; sc < Tkpad; sc += 32) {
        // stage V[sc..sc+31][0..HS) transposed -> Vs[n][k]
        for (int i = tid; i < 32 * HS; i += 256) {
            int k = i / HS, n = i - k * HS;       // consecutive tid -> consecutive n
            int s = sc + k;
            int scl = s < Tk ? s : Tk - 1;
            float v = Vh[(long long)scl * HS + n];
            if (s >= Tk) v = 0.f;
            Vs[n * 32 + k] = (_Float16)v;
        }
        __syncthreads();

        // P fragment for this key chunk (contiguous LDS reads)
        AFrag pa;
        {
            const int m = lane & 15;
            pa.v = ld16(&SP[m * Tkpad + sc + kbA]);
        }
        {
            const int n = wid * 16 + (lane & 15);
            AFrag vbf; vbf.v = ld16(&Vs[n * 32 + kbB]);
            oacc0 = wmma_f16(pa.v, vbf.v, oacc0);
        }
        if (ntiles > 8) {
            const int n = (wid + 8) * 16 + (lane & 15);
            AFrag vbf; vbf.v = ld16(&Vs[n * 32 + kbB]);
            oacc1 = wmma_f16(pa.v, vbf.v, oacc1);
        }
        __syncthreads();
    }

    // ---- store O (merged heads) ----
    {
        const int rb = 8 * (lane >> 4);
        const int n0 = wid * 16 + (lane & 15);
#pragma unroll
        for (int r = 0; r < 8; ++r) {
            int row = q0 + rb + r;
            if (row < Tq)
                O[(long long)(b * Tq + row) * ((long long)H * HS)
                  + (long long)h * HS + n0] = oacc0[r];
        }
        if (ntiles > 8) {
            const int n1 = (wid + 8) * 16 + (lane & 15);
#pragma unroll
            for (int r = 0; r < 8; ++r) {
                int row = q0 + rb + r;
                if (row < Tq)
                    O[(long long)(b * Tq + row) * ((long long)H * HS)
                      + (long long)h * HS + n1] = oacc1[r];
            }
        }
    }
}

// ---------------------------------------------------------------------------
// LayerNorm: one workgroup per row
// ---------------------------------------------------------------------------
__global__ void __launch_bounds__(256)
layernorm_k(const float* __restrict__ X, const float* __restrict__ g,
            const float* __restrict__ bta, float* __restrict__ Y, int D)
{
    __shared__ float red[256];
    const int row = blockIdx.x;
    const int tid = threadIdx.x;
    const float* x = X + (long long)row * D;
    float* y = Y + (long long)row * D;

    float s = 0.f;
    for (int i = tid; i < D; i += 256) s += x[i];
    red[tid] = s; __syncthreads();
    for (int off = 128; off > 0; off >>= 1) {
        if (tid < off) red[tid] += red[tid + off];
        __syncthreads();
    }
    const float mu = red[0] / (float)D;
    __syncthreads();

    float v = 0.f;
    for (int i = tid; i < D; i += 256) { float d = x[i] - mu; v += d * d; }
    red[tid] = v; __syncthreads();
    for (int off = 128; off > 0; off >>= 1) {
        if (tid < off) red[tid] += red[tid + off];
        __syncthreads();
    }
    const float inv = rsqrtf(red[0] / (float)D + 1e-5f);
    __syncthreads();

    for (int i = tid; i < D; i += 256)
        y[i] = (x[i] - mu) * inv * g[i] + bta[i];
}

// ---------------------------------------------------------------------------
// RoPE kernels. inv_j = theta^(-4j/hs) (faithful to _freqs_1d/_freqs_2d).
// X laid out as [rows][HS]; position t = row % T.
// ---------------------------------------------------------------------------
__global__ void rope1d_k(float* __restrict__ X, int T, int HS, long long rows)
{
    const int pairs = HS >> 1;
    const long long tot = rows * pairs;
    for (long long idx = blockIdx.x * (long long)blockDim.x + threadIdx.x;
         idx < tot; idx += (long long)gridDim.x * blockDim.x) {
        long long row = idx / pairs;
        int j = (int)(idx - row * pairs);
        int t = (int)(row % T);
        float ang = (float)t * __powf(10000.f, (-4.f * j) / (float)HS);
        float c = __cosf(ang), s = __sinf(ang);
        float* p = X + row * (long long)HS + 2 * j;
        float x1 = p[0], x2 = p[1];
        p[0] = x1 * c - x2 * s;
        p[1] = x1 * s + x2 * c;
    }
}

__global__ void rope2d_k(float* __restrict__ X, int T, int HS, int Wk, long long rows)
{
    const int hp = HS >> 2;                 // pairs per half
    const long long tot = rows * (HS >> 1);
    for (long long idx = blockIdx.x * (long long)blockDim.x + threadIdx.x;
         idx < tot; idx += (long long)gridDim.x * blockDim.x) {
        long long row = idx / (HS >> 1);
        int pj = (int)(idx - row * (HS >> 1));
        int half = pj / hp, j = pj - half * hp;
        int t = (int)(row % T);
        int pos = (half == 0) ? (t % Wk) : (t / Wk);
        float ang = (float)pos * __powf(10000.f, (-4.f * j) / (float)HS);
        float c = __cosf(ang), s = __sinf(ang);
        float* p = X + row * (long long)HS + half * (HS >> 1) + 2 * j;
        float x1 = p[0], x2 = p[1];
        p[0] = x1 * c - x2 * s;
        p[1] = x1 * s + x2 * c;
    }
}

// ---------------------------------------------------------------------------
// Token embedding gather
// ---------------------------------------------------------------------------
__global__ void embed_k(const int* __restrict__ tokens,
                        const float* __restrict__ table,
                        float* __restrict__ out, int n, int D)
{
    const long long tot = (long long)n * D;
    for (long long i = blockIdx.x * (long long)blockDim.x + threadIdx.x;
         i < tot; i += (long long)gridDim.x * blockDim.x) {
        long long r = i / D;
        int d = (int)(i - r * D);
        out[i] = table[(long long)tokens[r] * D + d];
    }
}

// ---------------------------------------------------------------------------
// Host orchestration
// ---------------------------------------------------------------------------
extern "C" void kernel_launch(void* const* d_in, const int* in_sizes, int n_in,
                              void* d_out, int out_size, void* d_ws, size_t ws_size,
                              hipStream_t stream)
{
    (void)in_sizes; (void)n_in; (void)out_size; (void)ws_size;

    const float* x_image    = (const float*)d_in[0];
    const int*   tokens     = (const int*)  d_in[1];
    const int*   latex_mask = (const int*)  d_in[2];
    const float* tok_table  = (const float*)d_in[3];
    const float* W_emb      = (const float*)d_in[4];
    const float* b_emb      = (const float*)d_in[5];
    const float* vit_ln1_g  = (const float*)d_in[6];
    const float* vit_ln1_b  = (const float*)d_in[7];
    const float* vit_wq     = (const float*)d_in[8];
    const float* vit_wk     = (const float*)d_in[9];
    const float* vit_wv     = (const float*)d_in[10];
    const float* vit_wproj  = (const float*)d_in[11];
    const float* vit_bproj  = (const float*)d_in[12];
    const float* vit_ln2_g  = (const float*)d_in[13];
    const float* vit_ln2_b  = (const float*)d_in[14];
    const float* vit_mw1    = (const float*)d_in[15];
    const float* vit_mb1    = (const float*)d_in[16];
    const float* vit_mw2    = (const float*)d_in[17];
    const float* vit_mb2    = (const float*)d_in[18];
    const float* lx_ln1_g   = (const float*)d_in[19];
    const float* lx_ln1_b   = (const float*)d_in[20];
    const float* lx_wq      = (const float*)d_in[21];
    const float* lx_wk      = (const float*)d_in[22];
    const float* lx_wv      = (const float*)d_in[23];
    const float* lx_wproj   = (const float*)d_in[24];
    const float* lx_bproj   = (const float*)d_in[25];
    const float* lx_ln2_g   = (const float*)d_in[26];
    const float* lx_ln2_b   = (const float*)d_in[27];
    const float* lx_mw1     = (const float*)d_in[28];
    const float* lx_mb1     = (const float*)d_in[29];
    const float* lx_mw2     = (const float*)d_in[30];
    const float* lx_mb2     = (const float*)d_in[31];
    const float* cr_ln1_g   = (const float*)d_in[32];
    const float* cr_ln1_b   = (const float*)d_in[33];
    const float* cr_ln2_g   = (const float*)d_in[34];
    const float* cr_ln2_b   = (const float*)d_in[35];
    const float* cr_wq      = (const float*)d_in[36];
    const float* cr_wk      = (const float*)d_in[37];
    const float* cr_wv      = (const float*)d_in[38];
    const float* cr_wproj   = (const float*)d_in[39];
    const float* cr_bproj   = (const float*)d_in[40];
    const float* cr_mw1     = (const float*)d_in[41];
    const float* cr_mb1     = (const float*)d_in[42];
    const float* cr_mw2     = (const float*)d_in[43];
    const float* cr_mb2     = (const float*)d_in[44];
    const float* hw1        = (const float*)d_in[45];
    const float* hb1        = (const float*)d_in[46];
    const float* hw2        = (const float*)d_in[47];
    const float* hb2        = (const float*)d_in[48];
    float* out = (float*)d_out;

    // workspace layout
    float* ws = (float*)d_ws;
    size_t off = 0;
    auto take = [&](size_t n) { float* p = ws + off; off += n; return p; };
    auto takeh = [&](size_t n) {           // f16 buffer, 16B aligned
        off = (off + 3) & ~(size_t)3;
        _Float16* p = (_Float16*)(ws + off);
        off += (n + 1) / 2;
        return p;
    };
    float* img = take((size_t)Bc * TIMG * DIMG);             // image stream
    float* xn  = take((size_t)Bc * TIMG * DIMG);             // LN scratch
    float* qb  = take((size_t)NHV * Bc * TIMG * HSV);        // Q [H][B][T][HS]
    float* kb  = take((size_t)NHV * Bc * TIMG * HSV);        // K
    float* vb  = take((size_t)NHV * Bc * TIMG * HSV);        // V
    float* ob  = take((size_t)Bc * TIMG * (size_t)NHV * HSV);// merged attn out
    float* h1  = take((size_t)Bc * TIMG * 4 * DIMG);         // MLP hidden
    float* txt = take((size_t)Bc * TTXT * DTXT);             // text stream
    float* tb1 = take((size_t)Bc * TTXT * DCR);              // text scratch A
    float* tb2 = take((size_t)Bc * TTXT * DCR);              // text scratch B

    // pre-transposed f16 weights [Np][Kp] (padded to 64x32 tiles)
    _Float16* wt_emb = takeh((size_t)DIMG * 416);
    _Float16* wt_vq  = takeh((size_t)2 * NHV * HSV * DIMG);
    _Float16* wt_vk  = takeh((size_t)2 * NHV * HSV * DIMG);
    _Float16* wt_vv  = takeh((size_t)2 * NHV * HSV * DIMG);
    _Float16* wt_vp  = takeh((size_t)2 * DIMG * NHV * HSV);
    _Float16* wt_vm1 = takeh((size_t)2 * 4 * DIMG * DIMG);
    _Float16* wt_vm2 = takeh((size_t)2 * DIMG * 4 * DIMG);
    _Float16* wt_lq  = takeh((size_t)NHL * HSL * DTXT);
    _Float16* wt_lk  = takeh((size_t)NHL * HSL * DTXT);
    _Float16* wt_lv  = takeh((size_t)NHL * HSL * DTXT);
    _Float16* wt_lp  = takeh((size_t)DTXT * NHL * HSL);
    _Float16* wt_lm1 = takeh((size_t)4 * DTXT * DTXT);
    _Float16* wt_lm2 = takeh((size_t)DTXT * 4 * DTXT);
    _Float16* wt_cq  = takeh((size_t)NHC * HSC * DTXT);
    _Float16* wt_ck  = takeh((size_t)NHC * HSC * DIMG);
    _Float16* wt_cv  = takeh((size_t)NHC * HSC * DIMG);
    _Float16* wt_cp  = takeh((size_t)DCR * NHC * HSC);
    _Float16* wt_cm1 = takeh((size_t)4 * DCR * DCR);
    _Float16* wt_cm2 = takeh((size_t)DCR * 4 * DCR);
    _Float16* wt_h1  = takeh((size_t)4 * DCR * DCR);
    _Float16* wt_h2  = takeh((size_t)320 * 4 * DCR);

    const float inv512 = 1.0f / sqrtf(512.f);
    const float inv128 = 1.0f / sqrtf(128.f);
    dim3 blk(256, 1, 1);

    auto wcvt = [&](const float* W, _Float16* Wt, int K, int N, int Kp, int Np,
                    int batch) {
        long long tot = (long long)Np * Kp;
        int blocks = (int)((tot + 255) / 256);
        if (blocks > 4096) blocks = 4096;
        dim3 g(blocks, 1, batch);
        wcvt_k<<<g, blk, 0, stream>>>(W, Wt, K, N, Kp, Np);
    };

    auto gemm = [&](const float* A, int lda, long long sA,
                    const _Float16* Bt, int kp, long long sB,
                    const float* bias, const float* Res, int ldres,
                    float* C, int ldc, long long sC,
                    int M, int N, int K, int relu, int batch) {
        dim3 grid((N + GT_N - 1) / GT_N, (M + GT_M - 1) / GT_M, batch);
        gemm_wmma_k<<<grid, blk, 0, stream>>>(A, lda, sA, Bt, kp, sB,
                                              bias, Res, ldres,
                                              C, ldc, sC, M, N, K, relu);
    };

    // ---- weight conversion (f32 [K,N] -> f16 [Np][Kp], transposed) ----
    wcvt(W_emb,    wt_emb, DPIC, DIMG, 416, DIMG, 1);
    wcvt(vit_wq,   wt_vq,  DIMG, HSV,  DIMG, HSV, 2 * NHV);
    wcvt(vit_wk,   wt_vk,  DIMG, HSV,  DIMG, HSV, 2 * NHV);
    wcvt(vit_wv,   wt_vv,  DIMG, HSV,  DIMG, HSV, 2 * NHV);
    wcvt(vit_wproj,wt_vp,  NHV * HSV, DIMG, NHV * HSV, DIMG, 2);
    wcvt(vit_mw1,  wt_vm1, DIMG, 4 * DIMG, DIMG, 4 * DIMG, 2);
    wcvt(vit_mw2,  wt_vm2, 4 * DIMG, DIMG, 4 * DIMG, DIMG, 2);
    wcvt(lx_wq,    wt_lq,  DTXT, HSL, DTXT, HSL, NHL);
    wcvt(lx_wk,    wt_lk,  DTXT, HSL, DTXT, HSL, NHL);
    wcvt(lx_wv,    wt_lv,  DTXT, HSL, DTXT, HSL, NHL);
    wcvt(lx_wproj, wt_lp,  NHL * HSL, DTXT, NHL * HSL, DTXT, 1);
    wcvt(lx_mw1,   wt_lm1, DTXT, 4 * DTXT, DTXT, 4 * DTXT, 1);
    wcvt(lx_mw2,   wt_lm2, 4 * DTXT, DTXT, 4 * DTXT, DTXT, 1);
    wcvt(cr_wq,    wt_cq,  DTXT, HSC, DTXT, HSC, NHC);
    wcvt(cr_wk,    wt_ck,  DIMG, HSC, DIMG, HSC, NHC);
    wcvt(cr_wv,    wt_cv,  DIMG, HSC, DIMG, HSC, NHC);
    wcvt(cr_wproj, wt_cp,  NHC * HSC, DCR, NHC * HSC, DCR, 1);
    wcvt(cr_mw1,   wt_cm1, DCR, 4 * DCR, DCR, 4 * DCR, 1);
    wcvt(cr_mw2,   wt_cm2, 4 * DCR, DCR, 4 * DCR, DCR, 1);
    wcvt(hw1,      wt_h1,  DCR, 4 * DCR, DCR, 4 * DCR, 1);
    wcvt(hw2,      wt_h2,  4 * DCR, VOC, 4 * DCR, 320, 1);

    // ---- image patch embedding ----
    gemm(x_image, DPIC, 0, wt_emb, 416, 0, b_emb, nullptr, 0,
         img, DIMG, 0, Bc * TIMG, DIMG, DPIC, 0, 1);

    // ---- 2 ResViT blocks ----
    for (int l = 0; l < 2; ++l) {
        const _Float16* wq = wt_vq + (size_t)l * NHV * HSV * DIMG;
        const _Float16* wk = wt_vk + (size_t)l * NHV * HSV * DIMG;
        const _Float16* wv = wt_vv + (size_t)l * NHV * HSV * DIMG;
        const _Float16* wp = wt_vp + (size_t)l * DIMG * NHV * HSV;

        layernorm_k<<<Bc * TIMG, blk, 0, stream>>>(img, vit_ln1_g + l * DIMG,
                                                   vit_ln1_b + l * DIMG, xn, DIMG);
        gemm(xn, DIMG, 0, wq, DIMG, (long long)HSV * DIMG, nullptr, nullptr, 0,
             qb, HSV, (long long)Bc * TIMG * HSV, Bc * TIMG, HSV, DIMG, 0, NHV);
        gemm(xn, DIMG, 0, wk, DIMG, (long long)HSV * DIMG, nullptr, nullptr, 0,
             kb, HSV, (long long)Bc * TIMG * HSV, Bc * TIMG, HSV, DIMG, 0, NHV);
        gemm(xn, DIMG, 0, wv, DIMG, (long long)HSV * DIMG, nullptr, nullptr, 0,
             vb, HSV, (long long)Bc * TIMG * HSV, Bc * TIMG, HSV, DIMG, 0, NHV);

        long long rows = (long long)NHV * Bc * TIMG;
        int nb = (int)((rows * (HSV / 2) + 255) / 256);
        rope2d_k<<<nb, blk, 0, stream>>>(qb, TIMG, HSV, 40, rows);
        rope2d_k<<<nb, blk, 0, stream>>>(kb, TIMG, HSV, 40, rows);

        dim3 ga(TIMG / 16, NHV, Bc);
        attn_wmma_k<<<ga, blk, 0, stream>>>(qb, kb, vb, ob, nullptr,
                                            Bc, NHV, TIMG, TIMG, HSV, inv512, 0);

        gemm(ob, NHV * HSV, 0, wp, NHV * HSV, 0, vit_bproj + l * DIMG, img, DIMG,
             img, DIMG, 0, Bc * TIMG, DIMG, NHV * HSV, 0, 1);
        layernorm_k<<<Bc * TIMG, blk, 0, stream>>>(img, vit_ln2_g + l * DIMG,
                                                   vit_ln2_b + l * DIMG, xn, DIMG);
        gemm(xn, DIMG, 0, wt_vm1 + (size_t)l * 4 * DIMG * DIMG, DIMG, 0,
             vit_mb1 + l * 4 * DIMG, nullptr, 0,
             h1, 4 * DIMG, 0, Bc * TIMG, 4 * DIMG, DIMG, 1, 1);
        gemm(h1, 4 * DIMG, 0, wt_vm2 + (size_t)l * DIMG * 4 * DIMG, 4 * DIMG, 0,
             vit_mb2 + l * DIMG, img, DIMG,
             img, DIMG, 0, Bc * TIMG, DIMG, 4 * DIMG, 0, 1);
    }

    // ---- latex self-attention block ----
    embed_k<<<(Bc * TTXT * DTXT + 255) / 256, blk, 0, stream>>>(tokens, tok_table,
                                                                txt, Bc * TTXT, DTXT);
    layernorm_k<<<Bc * TTXT, blk, 0, stream>>>(txt, lx_ln1_g, lx_ln1_b, tb2, DTXT);
    gemm(tb2, DTXT, 0, wt_lq, DTXT, (long long)HSL * DTXT, nullptr, nullptr, 0,
         qb, HSL, (long long)Bc * TTXT * HSL, Bc * TTXT, HSL, DTXT, 0, NHL);
    gemm(tb2, DTXT, 0, wt_lk, DTXT, (long long)HSL * DTXT, nullptr, nullptr, 0,
         kb, HSL, (long long)Bc * TTXT * HSL, Bc * TTXT, HSL, DTXT, 0, NHL);
    gemm(tb2, DTXT, 0, wt_lv, DTXT, (long long)HSL * DTXT, nullptr, nullptr, 0,
         vb, HSL, (long long)Bc * TTXT * HSL, Bc * TTXT, HSL, DTXT, 0, NHL);
    {
        long long rows = (long long)NHL * Bc * TTXT;
        int nb = (int)((rows * (HSL / 2) + 255) / 256);
        rope1d_k<<<nb, blk, 0, stream>>>(qb, TTXT, HSL, rows);
        rope1d_k<<<nb, blk, 0, stream>>>(kb, TTXT, HSL, rows);
    }
    {
        dim3 gl((TTXT + 15) / 16, NHL, Bc);
        attn_wmma_k<<<gl, blk, 0, stream>>>(qb, kb, vb, ob, latex_mask,
                                            Bc, NHL, TTXT, TTXT, HSL, inv128, 1);
    }
    gemm(ob, NHL * HSL, 0, wt_lp, NHL * HSL, 0, lx_bproj, txt, DTXT,
         tb1, DTXT, 0, Bc * TTXT, DTXT, NHL * HSL, 0, 1);        // x = txt + o@Wp+b
    layernorm_k<<<Bc * TTXT, blk, 0, stream>>>(tb1, lx_ln2_g, lx_ln2_b, tb2, DTXT);
    gemm(tb2, DTXT, 0, wt_lm1, DTXT, 0, lx_mb1, nullptr, 0,
         h1, 4 * DTXT, 0, Bc * TTXT, 4 * DTXT, DTXT, 1, 1);
    gemm(h1, 4 * DTXT, 0, wt_lm2, 4 * DTXT, 0, lx_mb2, txt, DTXT,
         txt, DTXT, 0, Bc * TTXT, DTXT, 4 * DTXT, 0, 1);          // txt = txt + mlp

    // ---- cross-attention block ----
    layernorm_k<<<Bc * TIMG, blk, 0, stream>>>(img, cr_ln1_g, cr_ln1_b, xn, DIMG);
    layernorm_k<<<Bc * TTXT, blk, 0, stream>>>(txt, cr_ln2_g, cr_ln2_b, tb2, DTXT);
    gemm(tb2, DTXT, 0, wt_cq, DTXT, (long long)HSC * DTXT, nullptr, nullptr, 0,
         qb, HSC, (long long)Bc * TTXT * HSC, Bc * TTXT, HSC, DTXT, 0, NHC);
    gemm(xn, DIMG, 0, wt_ck, DIMG, (long long)HSC * DIMG, nullptr, nullptr, 0,
         kb, HSC, (long long)Bc * TIMG * HSC, Bc * TIMG, HSC, DIMG, 0, NHC);
    gemm(xn, DIMG, 0, wt_cv, DIMG, (long long)HSC * DIMG, nullptr, nullptr, 0,
         vb, HSC, (long long)Bc * TIMG * HSC, Bc * TIMG, HSC, DIMG, 0, NHC);
    {
        long long rq = (long long)NHC * Bc * TTXT;
        rope1d_k<<<(int)((rq * (HSC / 2) + 255) / 256), blk, 0, stream>>>(qb, TTXT, HSC, rq);
        long long rk = (long long)NHC * Bc * TIMG;
        rope2d_k<<<(int)((rk * (HSC / 2) + 255) / 256), blk, 0, stream>>>(kb, TIMG, HSC, 40, rk);
    }
    {
        dim3 gc((TTXT + 15) / 16, NHC, Bc);
        attn_wmma_k<<<gc, blk, 0, stream>>>(qb, kb, vb, ob, nullptr,
                                            Bc, NHC, TTXT, TIMG, HSC, inv512, 0);
    }
    gemm(ob, NHC * HSC, 0, wt_cp, NHC * HSC, 0, cr_bproj, nullptr, 0,
         tb1, DCR, 0, Bc * TTXT, DCR, NHC * HSC, 0, 1);           // x
    gemm(tb1, DCR, 0, wt_cm1, DCR, 0, cr_mb1, nullptr, 0,
         h1, 4 * DCR, 0, Bc * TTXT, 4 * DCR, DCR, 1, 1);
    gemm(h1, 4 * DCR, 0, wt_cm2, 4 * DCR, 0, cr_mb2, tb1, DCR,
         tb1, DCR, 0, Bc * TTXT, DCR, 4 * DCR, 0, 1);             // x = x + mlp(x)

    // ---- output head ----
    gemm(tb1, DCR, 0, wt_h1, DCR, 0, hb1, nullptr, 0,
         h1, 4 * DCR, 0, Bc * TTXT, 4 * DCR, DCR, 1, 1);
    gemm(h1, 4 * DCR, 0, wt_h2, 4 * DCR, 0, hb2, nullptr, 0,
         out, VOC, 0, Bc * TTXT, VOC, 4 * DCR, 0, 1);
}
